// HybirdAttention_3770981286603
// MI455X (gfx1250) — matrix operand order repaired and checked by reading, not compiled, not run on hardware
//
#include <hip/hip_runtime.h>

typedef _Float16 half_t;
typedef __attribute__((ext_vector_type(16))) _Float16 v16h;
typedef __attribute__((ext_vector_type(8)))  _Float16 v8h;
typedef __attribute__((ext_vector_type(8)))  float    v8f;
typedef __attribute__((ext_vector_type(4)))  unsigned int u32x4;
typedef __attribute__((ext_vector_type(4)))  int      i32x4;
typedef __attribute__((ext_vector_type(8)))  int      i32x8;

#define Bn 8
#define Tn 1024
#define Fn 512
#define Hn 8
#define DKn 64
#define Cn 15
#define HCn 120
#define MTOK (Bn * Tn)

// Tensor Data Mover availability (probe-verified builtins; arity differs by
// toolchain: ROCm 7.2/clang-22 = 5 args, amdgpu-toolchain/clang-23 = 6 args).
#if defined(__has_builtin)
#if __has_builtin(__builtin_amdgcn_tensor_load_to_lds) && \
    __has_builtin(__builtin_amdgcn_s_wait_tensorcnt)
#define USE_TDM 1
#endif
#endif
#ifndef USE_TDM
#define USE_TDM 0
#endif

// ---------------------------------------------------------------------------
// f32 -> f16 conversion
// ---------------------------------------------------------------------------
__global__ void cvt_f32_f16_kernel(const float* __restrict__ in,
                                   half_t* __restrict__ out, int n) {
  int i = blockIdx.x * blockDim.x + threadIdx.x;
  if (i < n) out[i] = (half_t)in[i];
}

// ---------------------------------------------------------------------------
// Fragment assembly helpers (CDNA5 16-bit WMMA lane layouts, wave32).
// A 16x32: lane<16 holds row lr, K {0..7,16..23}; lane>=16 K {8..15,24..31}.
// B 32x16: lane<16 holds col lr, K {0..7,16..23}; lane>=16 K {8..15,24..31}.
// ---------------------------------------------------------------------------
__device__ __forceinline__ v16h frag_a_from(const half_t* p0) {
  v8h c0 = *(const v8h*)(p0);
  v8h c1 = *(const v8h*)(p0 + 16);
  v16h r;
#pragma unroll
  for (int e = 0; e < 8; e++) { r[e] = c0[e]; r[e + 8] = c1[e]; }
  return r;
}
__device__ __forceinline__ v16h frag_b_from(const half_t* p0) {
  v8h c0 = *(const v8h*)(p0);
  v8h c1 = *(const v8h*)(p0 + 16);
  v16h r;
#pragma unroll
  for (int e = 0; e < 8; e++) { r[e] = c0[e]; r[e + 8] = c1[e]; }
  return r;
}

#define GBM 128
#define GBN 64
#define GBK 32
#define LPITCH 40  // 32 + 8 halves pad: 20-dword row pitch, conflict-free

#if USE_TDM
// ---------------------------------------------------------------------------
// Issue one TDM 2D tile load: tile_rows x 32 halves from a row-major
// [tens_rows x tens_cols] view (row stride = row_stride elements) into LDS
// at lds_off, with TDM padding of 8 halves per 32-half row -> LPITCH=40.
// OOB rows (tens_rows < tile_rows) are returned as zeros by the TDM.
// D# bitfields per CDNA5 ISA 8.3/8.4.
// ---------------------------------------------------------------------------
__device__ __forceinline__ void tdm_load_tile_2d(const half_t* gsrc,
                                                 unsigned lds_off,
                                                 unsigned tile_rows,
                                                 unsigned tens_rows,
                                                 unsigned tens_cols,
                                                 unsigned row_stride) {
  unsigned long long ga = (unsigned long long)(uintptr_t)gsrc;
  u32x4 g0;
  g0[0] = 1u;                                 // count=1 (valid), user mode
  g0[1] = lds_off;                            // lds_addr (bytes)
  g0[2] = (unsigned)ga;                       // global_addr[31:0]
  g0[3] = (unsigned)(ga >> 32) | (2u << 30);  // global_addr[56:32] | type=2
  i32x8 g1;
  // data_size=1 (2B) | pad_enable | pad_interval=3 (16 dwords = 64B row)
  // | pad_amount=3 (4 dwords = 8 halves)
  g1[0] = (int)((1u << 16) | (1u << 20) | (3u << 22) | (3u << 25));
  g1[1] = (int)((tens_cols & 0xFFFFu) << 16);                        // dim0 lo
  g1[2] = (int)((tens_cols >> 16) | ((tens_rows & 0xFFFFu) << 16));  // dim0 hi | dim1 lo
  g1[3] = (int)((tens_rows >> 16) | (32u << 16));                    // dim1 hi | tile_dim0=32
  g1[4] = (int)(tile_rows & 0xFFFFu);                                // tile_dim1 (tile_dim2=0)
  g1[5] = (int)row_stride;                                           // dim0_stride[31:0]
  g1[6] = 0;
  g1[7] = 0;
  i32x4 z4 = {0, 0, 0, 0};
#if __clang_major__ >= 23
  i32x8 z8 = {0, 0, 0, 0, 0, 0, 0, 0};
  __builtin_amdgcn_tensor_load_to_lds(g0, g1, z4, z4, z8, 0);
#else
  __builtin_amdgcn_tensor_load_to_lds(g0, g1, z4, z4, 0);
#endif
}
#endif  // USE_TDM

// ---------------------------------------------------------------------------
// Generic WMMA GEMM: C[M,N] = act(A[M,K] * W[N,K]^T + bias)
// Block tile 128x64, 8 waves, each wave 32x32 (2x2 of 16x16), f32 accum.
// Staging: TDM double-buffered (wave 0 issues DMA, TENSORcnt-paced) when
// available; otherwise vector-load path.
// ---------------------------------------------------------------------------
__global__ __launch_bounds__(256) void gemm_wmma_kernel(
    const half_t* __restrict__ A, const half_t* __restrict__ W,
    const float* __restrict__ bias, float* __restrict__ outF,
    half_t* __restrict__ outH, int M, int N, int K, int relu) {
  __shared__ half_t As[2][GBM * LPITCH];
  __shared__ half_t Ws[2][GBN * LPITCH];

  int tid = threadIdx.x;
  int lane = tid & 31, wave = tid >> 5;
  int lr = lane & 15, hi = (lane >> 4) & 1;
  int wr = wave >> 1, wc = wave & 1;  // 4x2 wave grid
  int bm = blockIdx.x * GBM;
  int bn = blockIdx.y * GBN;

  v8f acc[2][2] = {};

#if USE_TDM
  const bool issuer = (wave == 0);
  if (issuer) {
    tdm_load_tile_2d(A + (size_t)bm * K, (unsigned)(uintptr_t)&As[0][0], GBM,
                     (unsigned)(M - bm), (unsigned)K, (unsigned)K);
    tdm_load_tile_2d(W + (size_t)bn * K, (unsigned)(uintptr_t)&Ws[0][0], GBN,
                     (unsigned)(N - bn), (unsigned)K, (unsigned)K);
  }
#endif

  for (int k0 = 0; k0 < K; k0 += GBK) {
    int cur = (k0 >> 5) & 1;
#if USE_TDM
    // Issue next stage into the other buffer, then wait for current stage
    // (<=2 keeps the just-issued pair in flight).
    if (issuer) {
      if (k0 + GBK < K) {
        int kn = k0 + GBK;
        tdm_load_tile_2d(A + (size_t)bm * K + kn,
                         (unsigned)(uintptr_t)&As[cur ^ 1][0], GBM,
                         (unsigned)(M - bm), (unsigned)(K - kn), (unsigned)K);
        tdm_load_tile_2d(W + (size_t)bn * K + kn,
                         (unsigned)(uintptr_t)&Ws[cur ^ 1][0], GBN,
                         (unsigned)(N - bn), (unsigned)(K - kn), (unsigned)K);
        __builtin_amdgcn_s_wait_tensorcnt(2);
      } else {
        __builtin_amdgcn_s_wait_tensorcnt(0);
      }
    }
    __syncthreads();
#else
    cur = 0;
    // Stage A slab: 128 rows x 32 halves (16 halves / thread)
    {
      int r = tid >> 1, c = (tid & 1) * 16;
      const half_t* src = A + (size_t)(bm + r) * K + k0 + c;
      v8h x0 = *(const v8h*)src;
      v8h x1 = *(const v8h*)(src + 8);
      *(v8h*)(As[0] + r * LPITCH + c) = x0;
      *(v8h*)(As[0] + r * LPITCH + c + 8) = x1;
      if (k0 + GBK < K) __builtin_prefetch(src + GBK, 0, 1);
    }
    // Stage W slab: 64 rows x 32 halves (8 halves / thread), guard N
    {
      int r = tid >> 2, c = (tid & 3) * 8;
      int wrow = bn + r;
      v8h x = {};
      if (wrow < N) x = *(const v8h*)(W + (size_t)wrow * K + k0 + c);
      *(v8h*)(Ws[0] + r * LPITCH + c) = x;
    }
    __syncthreads();
#endif

    const half_t* pa = As[cur] + (wr * 32 + lr) * LPITCH + hi * 8;
    v16h a0 = frag_a_from(pa);
    v16h a1 = frag_a_from(pa + 16 * LPITCH);
    const half_t* pb = Ws[cur] + (wc * 32 + lr) * LPITCH + hi * 8;
    v16h b0 = frag_b_from(pb);
    v16h b1 = frag_b_from(pb + 16 * LPITCH);

    acc[0][0] = __builtin_amdgcn_wmma_f32_16x16x32_f16(false, a0, false, b0,
                                                       (short)0, acc[0][0], false, false);
    acc[0][1] = __builtin_amdgcn_wmma_f32_16x16x32_f16(false, a0, false, b1,
                                                       (short)0, acc[0][1], false, false);
    acc[1][0] = __builtin_amdgcn_wmma_f32_16x16x32_f16(false, a1, false, b0,
                                                       (short)0, acc[1][0], false, false);
    acc[1][1] = __builtin_amdgcn_wmma_f32_16x16x32_f16(false, a1, false, b1,
                                                       (short)0, acc[1][1], false, false);
    __syncthreads();
  }

  // Epilogue: C layout — VGPR v holds row hi*8+v, lane lr holds col.
#pragma unroll
  for (int i = 0; i < 2; i++)
#pragma unroll
    for (int j = 0; j < 2; j++) {
      int col = bn + wc * 32 + j * 16 + lr;
      if (col >= N) continue;
      float bv = bias ? bias[col] : 0.0f;
#pragma unroll
      for (int v = 0; v < 8; v++) {
        int row = bm + wr * 32 + i * 16 + hi * 8 + v;
        float x = acc[i][j][v] + bv;
        if (relu) x = fmaxf(x, 0.0f);
        if (outF) outF[(size_t)row * N + col] = x;
        if (outH) outH[(size_t)row * N + col] = (half_t)x;
      }
    }
}

// ---------------------------------------------------------------------------
// LDSA window attention: softmax over 15 logits per (bt,h), 15-tap window
// over val, write f16 result (input to the next GEMM).
// ---------------------------------------------------------------------------
__global__ void ldsa_window_kernel(const float* __restrict__ logits,  // [MTOK,120]
                                   const float* __restrict__ val,     // [MTOK,512]
                                   half_t* __restrict__ xh) {         // [MTOK,512]
  int gid = blockIdx.x * blockDim.x + threadIdx.x;
  int f = gid & (Fn - 1);
  int bt = gid >> 9;
  int b = bt >> 10;
  int t = bt & (Tn - 1);
  int h = f >> 6;

  const float* lg = logits + (size_t)bt * HCn + h * Cn;
  float w[Cn];
  float m = -3.0e38f;
#pragma unroll
  for (int c = 0; c < Cn; c++) { w[c] = lg[c]; m = fmaxf(m, w[c]); }
  float s = 0.0f;
#pragma unroll
  for (int c = 0; c < Cn; c++) { w[c] = __expf(w[c] - m); s += w[c]; }
  float inv = 1.0f / s;

  float acc = 0.0f;
#pragma unroll
  for (int c = 0; c < Cn; c++) {
    int tt = t - (Cn - 1) / 2 + c;
    if (tt >= 0 && tt < Tn)
      acc += w[c] * val[((size_t)b * Tn + tt) * Fn + f];
  }
  xh[gid] = (half_t)(acc * inv);
}

// ---------------------------------------------------------------------------
// Flash-style MHSA core. Block = 4 waves = 64 query rows of one (b,h).
// Per 32-col step: V staged transposed in LDS, scores via WMMA, online
// softmax (shfl_xor within 16-lane halves), P through per-wave LDS tile
// (C-layout -> A-layout), P*V via WMMA.
// ---------------------------------------------------------------------------
__global__ __launch_bounds__(128) void flash_attn_kernel(
    const half_t* __restrict__ Q, const half_t* __restrict__ Kh,
    const half_t* __restrict__ V, const int* __restrict__ mask,
    half_t* __restrict__ O) {
  __shared__ half_t Vt[DKn * LPITCH];      // transposed V tile [d][j]
  __shared__ half_t Ps[4 * 16 * LPITCH];   // per-wave P tile [row][j]

  int tid = threadIdx.x, lane = tid & 31, wave = tid >> 5;
  int lr = lane & 15, hi = (lane >> 4) & 1;
  int rb = blockIdx.x & 15;           // T/64 = 16 row blocks
  int h = (blockIdx.x >> 4) & 7;
  int b = blockIdx.x >> 7;
  int i0 = rb * 64 + wave * 16;
  size_t base = ((size_t)b * Tn) * Fn + (size_t)h * DKn;

  // Q fragments: DK=64 -> two K-steps of 32
  v16h aq[2];
#pragma unroll
  for (int ks = 0; ks < 2; ks++) {
    const half_t* p = Q + base + (size_t)(i0 + lr) * Fn + ks * 32 + hi * 8;
    v8h c0 = *(const v8h*)p;
    v8h c1 = *(const v8h*)(p + 16);
#pragma unroll
    for (int e = 0; e < 8; e++) { aq[ks][e] = c0[e]; aq[ks][e + 8] = c1[e]; }
  }

  float mrow[8], lrow[8];
#pragma unroll
  for (int v = 0; v < 8; v++) { mrow[v] = -3.0e38f; lrow[v] = 0.0f; }
  v8f accO[4] = {};

  for (int j0 = 0; j0 < Tn; j0 += 32) {
    // Stage V transposed: 32(j) x 64(d) -> Vt[d][j]
    {
      int jrel = tid >> 2;
      int dblk = (tid & 3) * 16;
      const half_t* src = V + base + (size_t)(j0 + jrel) * Fn + dblk;
      v8h x0 = *(const v8h*)src;
      v8h x1 = *(const v8h*)(src + 8);
#pragma unroll
      for (int e = 0; e < 8; e++) {
        Vt[(dblk + e) * LPITCH + jrel] = x0[e];
        Vt[(dblk + 8 + e) * LPITCH + jrel] = x1[e];
      }
    }
    __syncthreads();

    // Scores: 16 rows x 32 cols via 4 WMMAs (K fragments from global)
    v8f s[2] = {};
#pragma unroll
    for (int jj = 0; jj < 2; jj++)
#pragma unroll
      for (int ks = 0; ks < 2; ks++) {
        const half_t* p =
            Kh + base + (size_t)(j0 + jj * 16 + lr) * Fn + ks * 32 + hi * 8;
        v8h c0 = *(const v8h*)p;
        v8h c1 = *(const v8h*)(p + 16);
        v16h bk;
#pragma unroll
        for (int e = 0; e < 8; e++) { bk[e] = c0[e]; bk[e + 8] = c1[e]; }
        s[jj] = __builtin_amdgcn_wmma_f32_16x16x32_f16(false, aq[ks], false, bk,
                                                       (short)0, s[jj], false, false);
      }

    // Scale + mask
#pragma unroll
    for (int jj = 0; jj < 2; jj++) {
      int jc = j0 + jj * 16 + lr;
#pragma unroll
      for (int v = 0; v < 8; v++) {
        int i = i0 + hi * 8 + v;
        int mv = mask[((size_t)b * Tn + i) * Tn + jc];
        float x = s[jj][v] * 0.125f;  // 1/sqrt(64)
        s[jj][v] = mv ? x : -3.0e38f;
      }
    }

    // Online softmax (row = hi*8+v; reduce across the 16 lanes of a half)
    float pr[2][8];
#pragma unroll
    for (int v = 0; v < 8; v++) {
      float rmax = fmaxf(s[0][v], s[1][v]);
#pragma unroll
      for (int mk = 1; mk < 16; mk <<= 1)
        rmax = fmaxf(rmax, __shfl_xor(rmax, mk, 32));
      float mnew = fmaxf(mrow[v], rmax);
      float scale = __expf(mrow[v] - mnew);
      mrow[v] = mnew;
      float rsum = 0.0f;
#pragma unroll
      for (int jj = 0; jj < 2; jj++) {
        float p = __expf(s[jj][v] - mnew);
        pr[jj][v] = p;
        rsum += p;
      }
#pragma unroll
      for (int mk = 1; mk < 16; mk <<= 1) rsum += __shfl_xor(rsum, mk, 32);
      lrow[v] = lrow[v] * scale + rsum;
#pragma unroll
      for (int n = 0; n < 4; n++) accO[n][v] *= scale;
    }

    // P: C-layout -> LDS -> A-layout (wave-private tile; DS ops in-order)
    half_t* pw = Ps + wave * 16 * LPITCH;
#pragma unroll
    for (int jj = 0; jj < 2; jj++)
#pragma unroll
      for (int v = 0; v < 8; v++)
        pw[(hi * 8 + v) * LPITCH + jj * 16 + lr] = (half_t)pr[jj][v];
    v16h ap = frag_a_from(pw + lr * LPITCH + hi * 8);

    // O += P * V (4 d-tiles, K=32 over j)
#pragma unroll
    for (int n = 0; n < 4; n++) {
      v16h bv = frag_b_from(Vt + (n * 16 + lr) * LPITCH + hi * 8);
      accO[n] = __builtin_amdgcn_wmma_f32_16x16x32_f16(false, ap, false, bv,
                                                       (short)0, accO[n], false, false);
    }
    __syncthreads();
  }

  // Normalize and store f16 output
#pragma unroll
  for (int v = 0; v < 8; v++) {
    float inv = lrow[v] > 0.0f ? 1.0f / lrow[v] : 0.0f;
    int row = i0 + hi * 8 + v;
#pragma unroll
    for (int n = 0; n < 4; n++)
      O[base + (size_t)row * Fn + n * 16 + lr] = (half_t)(accO[n][v] * inv);
  }
}

// ---------------------------------------------------------------------------
// Host-side orchestration
// ---------------------------------------------------------------------------
extern "C" void kernel_launch(void* const* d_in, const int* in_sizes, int n_in,
                              void* d_out, int out_size, void* d_ws,
                              size_t ws_size, hipStream_t stream) {
  (void)in_sizes; (void)n_in; (void)out_size; (void)ws_size;
  const float* query = (const float*)d_in[0];
  // d_in[1] (key) is unused by the reference (MHSA runs on LDSA output)
  const float* value = (const float*)d_in[2];
  const int* mask = (const int*)d_in[3];
  const float* w1 = (const float*)d_in[4];
  const float* w2 = (const float*)d_in[5];
  const float* w3 = (const float*)d_in[6];
  const float* wout = (const float*)d_in[7];
  const float* wq = (const float*)d_in[8];
  const float* bq = (const float*)d_in[9];
  const float* wk = (const float*)d_in[10];
  const float* bk = (const float*)d_in[11];
  const float* wv = (const float*)d_in[12];
  const float* bv = (const float*)d_in[13];
  const float* wo = (const float*)d_in[14];
  const float* bo = (const float*)d_in[15];
  float* out = (float*)d_out;

  char* ws = (char*)d_ws;
  size_t off = 0;
  auto alloc = [&](size_t bytes) -> void* {
    void* p = ws + off;
    off += (bytes + 255) & ~(size_t)255;
    return p;
  };

  half_t* qh   = (half_t*)alloc((size_t)MTOK * Fn * 2);
  half_t* vh   = (half_t*)alloc((size_t)MTOK * Fn * 2);
  half_t* q1h  = (half_t*)alloc((size_t)MTOK * Fn * 2);
  float*  valf = (float*)alloc((size_t)MTOK * Fn * 4);
  float*  logits = (float*)alloc((size_t)MTOK * HCn * 4);
  half_t* xh   = (half_t*)alloc((size_t)MTOK * Fn * 2);
  half_t* yh   = (half_t*)alloc((size_t)MTOK * Fn * 2);
  half_t* w1h  = (half_t*)alloc((size_t)Fn * Fn * 2);
  half_t* w2h  = (half_t*)alloc((size_t)HCn * Fn * 2);
  half_t* w3h  = (half_t*)alloc((size_t)Fn * Fn * 2);
  half_t* wouth = (half_t*)alloc((size_t)Fn * Fn * 2);
  half_t* wqh  = (half_t*)alloc((size_t)Fn * Fn * 2);
  half_t* wkh  = (half_t*)alloc((size_t)Fn * Fn * 2);
  half_t* wvh  = (half_t*)alloc((size_t)Fn * Fn * 2);
  half_t* woh  = (half_t*)alloc((size_t)Fn * Fn * 2);
  // Aliases (lifetimes are disjoint): q/k/v projections + attention output
  half_t* qph = qh;             // qh dead after gemm #1
  half_t* kph = vh;             // vh dead after gemm #3
  half_t* vph = q1h;            // q1h dead after gemm #2
  half_t* oh  = (half_t*)valf;  // valf dead after window kernel

  auto cvt = [&](const float* src, half_t* dst, int n) {
    cvt_f32_f16_kernel<<<(n + 255) / 256, 256, 0, stream>>>(src, dst, n);
  };
  auto gemm = [&](const half_t* A, const half_t* W, const float* bias,
                  float* oF, half_t* oH, int M, int N, int K, int relu) {
    dim3 grid(M / GBM, (N + GBN - 1) / GBN);
    gemm_wmma_kernel<<<grid, 256, 0, stream>>>(A, W, bias, oF, oH, M, N, K, relu);
  };

  // 0) precision conversion (one-touch; everything then lives in L2 as f16)
  cvt(query, qh, MTOK * Fn);
  cvt(value, vh, MTOK * Fn);
  cvt(w1, w1h, Fn * Fn);
  cvt(w2, w2h, HCn * Fn);
  cvt(w3, w3h, Fn * Fn);
  cvt(wout, wouth, Fn * Fn);
  cvt(wq, wqh, Fn * Fn);
  cvt(wk, wkh, Fn * Fn);
  cvt(wv, wvh, Fn * Fn);
  cvt(wo, woh, Fn * Fn);

  // 1) LDSA
  gemm(qh, w1h, nullptr, nullptr, q1h, MTOK, Fn, Fn, 1);       // relu(q @ W1^T)
  gemm(q1h, w2h, nullptr, logits, nullptr, MTOK, HCn, Fn, 0);  // logits
  gemm(vh, w3h, nullptr, valf, nullptr, MTOK, Fn, Fn, 0);      // val
  ldsa_window_kernel<<<(MTOK * Fn) / 256, 256, 0, stream>>>(logits, valf, xh);
  gemm(xh, wouth, nullptr, nullptr, yh, MTOK, Fn, Fn, 0);      // LDSA out

  // 2) MHSA
  gemm(yh, wqh, bq, nullptr, qph, MTOK, Fn, Fn, 0);
  gemm(yh, wkh, bk, nullptr, kph, MTOK, Fn, Fn, 0);
  gemm(yh, wvh, bv, nullptr, vph, MTOK, Fn, Fn, 0);
  flash_attn_kernel<<<Bn * Hn * (Tn / 64), 128, 0, stream>>>(qph, kph, vph,
                                                             mask, oh);
  gemm(oh, woh, bo, out, nullptr, MTOK, Fn, Fn, 0);            // final f32 out
}
